// SelfAttention2D_44178033607173
// MI455X (gfx1250) — matrix operand order repaired and checked by reading, not compiled, not run on hardware
//
#include <hip/hip_runtime.h>
#include <hip/hip_bf16.h>

// SelfAttention2D: qkv = conv3x3(x); scores = q^T k; attn = softmax; out = attn @ v^T
// B=8, C=64, H=32, W=64, N=2048.  bf16 WMMA (16x16x32) everywhere, f32 accumulate,
// flash-attention style online softmax, K/V tiles staged to LDS with gfx1250
// async global->LDS loads, double buffered on ASYNCcnt.

#define BATCH 8
#define CCH   64
#define HH    32
#define WW    64
#define NPOS  2048          // H*W
#define K9    576           // C*9 (im2col depth)

typedef __attribute__((ext_vector_type(16))) __bf16        v16bf;
typedef __attribute__((ext_vector_type(8)))  float         v8f;
typedef __attribute__((ext_vector_type(4)))  unsigned int  v4u;
typedef __attribute__((ext_vector_type(4)))  int           v4i;

union FragU { v16bf f; struct { v4u lo; v4u hi; } u; };

// Load a 32-byte WMMA fragment as two 128-bit global/LDS loads.
__device__ __forceinline__ v16bf load_frag(const __bf16* p0, const __bf16* p1) {
    FragU t;
    t.u.lo = *(const v4u*)p0;
    t.u.hi = *(const v4u*)p1;
    return t.f;
}

__device__ __forceinline__ v8f wmma_bf16(v16bf a, v16bf b, v8f c) {
    return __builtin_amdgcn_wmma_f32_16x16x32_bf16(false, a, false, b, (short)0, c,
                                                   false, false);
}

// ---- gfx1250 async global->LDS copy (ASYNCcnt-tracked) ---------------------
typedef __attribute__((address_space(1))) v4i* gptr128;
typedef __attribute__((address_space(3))) v4i* lptr128;

__device__ __forceinline__ void async_ld_b128(const __bf16* g, __bf16* l) {
#if defined(__has_builtin) && __has_builtin(__builtin_amdgcn_global_load_async_to_lds_b128)
    __builtin_amdgcn_global_load_async_to_lds_b128((gptr128)g, (lptr128)l, 0, 0);
#else
    unsigned loff = (unsigned)(size_t)(__attribute__((address_space(3))) char*)l;
    asm volatile("global_load_async_to_lds_b128 %0, %1, off"
                 :: "v"(loff), "v"(g) : "memory");
#endif
}

__device__ __forceinline__ void wait_async0() {
#if defined(__has_builtin) && __has_builtin(__builtin_amdgcn_s_wait_asynccnt)
    __builtin_amdgcn_s_wait_asynccnt(0);
#else
    asm volatile("s_wait_asynccnt 0x0" ::: "memory");
#endif
}

// ---------------------------------------------------------------------------
// Pack conv weights f32 [64][64][3][3] -> bf16 [3][64][576]  (row-major over k)
// ---------------------------------------------------------------------------
__global__ void pack_weights(const float* __restrict__ wq,
                             const float* __restrict__ wk,
                             const float* __restrict__ wv,
                             __bf16* __restrict__ Wb) {
    int idx = blockIdx.x * blockDim.x + threadIdx.x;
    const int total = 3 * CCH * K9;
    if (idx >= total) return;
    int k    = idx % K9;
    int cout = (idx / K9) % CCH;
    int s    = idx / (K9 * CCH);
    int cin  = k / 9;
    int r9   = k % 9;
    const float* w = (s == 0) ? wq : (s == 1) ? wk : wv;
    float v = w[((cout * CCH + cin) * 3 + (r9 / 3)) * 3 + (r9 % 3)];
    Wb[idx] = (__bf16)v;
}

// ---------------------------------------------------------------------------
// im2col with SAME padding: x f32 [B][C][H][W] -> Xcol bf16 [B][N][576]
// ---------------------------------------------------------------------------
__global__ void im2col_pack(const float* __restrict__ x, __bf16* __restrict__ Xcol) {
    size_t idx = (size_t)blockIdx.x * blockDim.x + threadIdx.x;
    const size_t total = (size_t)BATCH * NPOS * K9;
    if (idx >= total) return;
    int k = idx % K9;
    int n = (idx / K9) % NPOS;
    int b = idx / ((size_t)K9 * NPOS);
    int cin = k / 9;
    int r9  = k % 9;
    int ky = r9 / 3, kx = r9 % 3;
    int y = n / WW, xx = n % WW;
    int sy = y + ky - 1, sx = xx + kx - 1;
    float v = 0.0f;
    if (sy >= 0 && sy < HH && sx >= 0 && sx < WW)
        v = x[(((size_t)b * CCH + cin) * HH + sy) * WW + sx];
    Xcol[idx] = (__bf16)v;
}

// ---------------------------------------------------------------------------
// QKV GEMM: out[cout][n] = sum_k Wb[s][cout][k] * Xcol[b][n][k] + bias
// One wave per 16x16 output tile. 18 k-chunks of 32 -> 18 WMMA.
// Q,K stored n-major [b][n][c], V stored c-major [b][c][n].
// ---------------------------------------------------------------------------
__global__ __launch_bounds__(128)
void qkv_gemm(const __bf16* __restrict__ Wb, const __bf16* __restrict__ Xcol,
              const float* __restrict__ bq, const float* __restrict__ bk,
              const float* __restrict__ bv,
              __bf16* __restrict__ Qb, __bf16* __restrict__ Kb,
              __bf16* __restrict__ Vb) {
    int wid  = blockIdx.x * 4 + (threadIdx.x >> 5);
    int lane = threadIdx.x & 31;
    int lh   = lane & 15;
    bool hiL = lane >= 16;

    int nt = wid & 127;          // n tile (128)
    int mt = (wid >> 7) & 3;     // cout tile (4)
    int b  = (wid >> 9) & 7;     // batch (8)
    int s  = wid >> 12;          // 0=q 1=k 2=v
    int n0 = nt * 16, m0 = mt * 16;

    const __bf16* Wrow = Wb + ((size_t)(s * CCH) + m0 + lh) * K9;
    const __bf16* Xrow = Xcol + ((size_t)(b * NPOS) + n0 + lh) * K9;

    v8f acc = {};
#pragma unroll 3
    for (int kc = 0; kc < 18; ++kc) {
        int ka = kc * 32 + (hiL ? 8 : 0);    // A: K = {ka..ka+7, ka+16..ka+23}
        int kb = kc * 32 + (hiL ? 8 : 0);    // B: K = {kb..kb+7, kb+16..kb+23}
        v16bf a  = load_frag(Wrow + ka, Wrow + ka + 16);
        v16bf bf = load_frag(Xrow + kb, Xrow + kb + 16);
        acc = wmma_bf16(a, bf, acc);
    }

    const float* bias = (s == 0) ? bq : (s == 1) ? bk : bv;
#pragma unroll
    for (int r = 0; r < 8; ++r) {
        int cout = m0 + r + (hiL ? 8 : 0);
        int n    = n0 + lh;
        __bf16 v = (__bf16)(acc[r] + bias[cout]);
        if (s == 2)
            Vb[((size_t)(b * CCH) + cout) * NPOS + n] = v;
        else if (s == 0)
            Qb[((size_t)(b * NPOS) + n) * CCH + cout] = v;
        else
            Kb[((size_t)(b * NPOS) + n) * CCH + cout] = v;
    }
}

// ---------------------------------------------------------------------------
// Flash attention: 4 waves/block, one wave per 16-query tile; all 4 waves of a
// block share the same batch and walk the same K/V stream, so K/V 32-key tiles
// are staged into LDS once per block via async global->LDS loads (double
// buffered on ASYNCcnt), then each wave reads its WMMA B-fragments from LDS.
// ---------------------------------------------------------------------------
__global__ __launch_bounds__(128)
void flash_attn(const __bf16* __restrict__ Qb, const __bf16* __restrict__ Kb,
                const __bf16* __restrict__ Vb, float* __restrict__ out) {
    __shared__ __align__(16) __bf16 kbuf[2][32 * CCH];   // [j-local][c]   4KB each
    __shared__ __align__(16) __bf16 vbuf[2][CCH * 32];   // [c][j-local]   4KB each
    __shared__ __align__(16) __bf16 plds[4][16 * 48];    // P transpose, 48-el stride

    int tid   = threadIdx.x;
    int wslot = tid >> 5;
    int wid   = blockIdx.x * 4 + wslot;
    int lane  = tid & 31;
    int lh    = lane & 15;
    bool hiL  = lane >= 16;
    int qt = wid & 127;
    int b  = wid >> 7;          // uniform across the block
    int i0 = qt * 16;

    const __bf16* Krow  = Kb + (size_t)(b * NPOS) * CCH;
    const __bf16* Vbase = Vb + (size_t)(b * CCH) * NPOS;

    // Q A-fragments for channel chunks 0..31 and 32..63
    const __bf16* Qrow = Qb + ((size_t)(b * NPOS) + i0 + lh) * CCH;
    int ao = hiL ? 8 : 0;
    v16bf aq0 = load_frag(Qrow + ao,      Qrow + ao + 16);
    v16bf aq1 = load_frag(Qrow + 32 + ao, Qrow + 32 + ao + 16);

    float m[8], l[8];
    v8f O0 = {}, O1 = {}, O2 = {}, O3 = {};
#pragma unroll
    for (int r = 0; r < 8; ++r) { m[r] = -3.0e38f; l[r] = 0.0f; }

    __bf16* pt = plds[wslot];
    int cb = hiL ? 8 : 0;    // K-matrix B-frag channel offset
    int jb = hiL ? 8 : 0;    // V-matrix B-frag j offset

    // Stage K/V tile for key block jt into LDS buffer buf (128 threads).
    auto stage = [&](int jt_n, int buf) {
        int j0n = jt_n * 32;
        const __bf16* gk = Krow + (size_t)j0n * CCH;   // contiguous 4KB
        __bf16* lk = kbuf[buf];
#pragma unroll
        for (int p = 0; p < 2; ++p) {
            int e = (p * 128 + tid) * 8;               // 16B per thread per pass
            async_ld_b128(gk + e, lk + e);
        }
        __bf16* lv = vbuf[buf];
#pragma unroll
        for (int p = 0; p < 2; ++p) {
            int t = p * 128 + tid;
            int row = t >> 2, q = (t & 3) * 8;         // 4 threads per channel row
            async_ld_b128(Vbase + (size_t)row * NPOS + j0n + q, lv + row * 32 + q);
        }
    };

    stage(0, 0);

#pragma unroll 1
    for (int jt = 0; jt < 64; ++jt) {
        wait_async0();
        __syncthreads();                  // buffer jt&1 fully staged for all waves
        if (jt < 63) stage(jt + 1, (jt + 1) & 1);
        const __bf16* kb = kbuf[jt & 1];
        const __bf16* vb = vbuf[jt & 1];

        // ---- scores: S = Q^T K over this 32-key tile (B-frags from LDS) ----
        const __bf16* k0 = kb + (size_t)lh * CCH;          // j-local = lh
        const __bf16* k1 = kb + (size_t)(16 + lh) * CCH;   // j-local = 16+lh
        v16bf b00 = load_frag(k0 + cb,      k0 + cb + 16);
        v16bf b01 = load_frag(k0 + 32 + cb, k0 + 32 + cb + 16);
        v16bf b10 = load_frag(k1 + cb,      k1 + cb + 16);
        v16bf b11 = load_frag(k1 + 32 + cb, k1 + 32 + cb + 16);

        v8f S0 = {}, S1 = {};
        S0 = wmma_bf16(aq0, b00, S0);
        S0 = wmma_bf16(aq1, b01, S0);
        S1 = wmma_bf16(aq0, b10, S1);
        S1 = wmma_bf16(aq1, b11, S1);

        // ---- online softmax (row = r + 8*hiL; cols live in 16-lane halves) ----
        float scale[8];
#pragma unroll
        for (int r = 0; r < 8; ++r) {
            float t = fmaxf(S0[r], S1[r]);
            for (int off = 1; off < 16; off <<= 1) t = fmaxf(t, __shfl_xor(t, off, 32));
            float mn = fmaxf(m[r], t);
            scale[r] = __expf(m[r] - mn);
            float p0 = __expf(S0[r] - mn);
            float p1 = __expf(S1[r] - mn);
            S0[r] = p0; S1[r] = p1;
            float rs = p0 + p1;
            for (int off = 1; off < 16; off <<= 1) rs += __shfl_xor(rs, off, 32);
            l[r] = l[r] * scale[r] + rs;
            m[r] = mn;
        }

        // ---- transpose P through LDS into A-fragment order ----
#pragma unroll
        for (int r = 0; r < 8; ++r) {
            int row = r + (hiL ? 8 : 0);
            pt[row * 48 + lh]      = (__bf16)S0[r];
            pt[row * 48 + lh + 16] = (__bf16)S1[r];
        }
        asm volatile("s_wait_dscnt 0" ::: "memory");
        const __bf16* pr = pt + lh * 48 + (hiL ? 8 : 0);
        v16bf pa = load_frag(pr, pr + 16);

        // ---- rescale accumulators, then O += P @ V^T (B-frags from LDS) ----
#pragma unroll
        for (int r = 0; r < 8; ++r) {
            O0[r] *= scale[r]; O1[r] *= scale[r];
            O2[r] *= scale[r]; O3[r] *= scale[r];
        }
        const __bf16* v0 = vb + (size_t)(0  + lh) * 32 + jb;
        const __bf16* v1 = vb + (size_t)(16 + lh) * 32 + jb;
        const __bf16* v2 = vb + (size_t)(32 + lh) * 32 + jb;
        const __bf16* v3 = vb + (size_t)(48 + lh) * 32 + jb;
        O0 = wmma_bf16(pa, load_frag(v0, v0 + 16), O0);
        O1 = wmma_bf16(pa, load_frag(v1, v1 + 16), O1);
        O2 = wmma_bf16(pa, load_frag(v2, v2 + 16), O2);
        O3 = wmma_bf16(pa, load_frag(v3, v3 + 16), O3);
    }

    // ---- epilogue: out[b][c][i] = O[i][c] / l[i] ----
#pragma unroll
    for (int r = 0; r < 8; ++r) {
        float inv = 1.0f / l[r];
        size_t i = (size_t)i0 + r + (hiL ? 8 : 0);
        out[((size_t)(b * CCH) + 0  + lh) * NPOS + i] = O0[r] * inv;
        out[((size_t)(b * CCH) + 16 + lh) * NPOS + i] = O1[r] * inv;
        out[((size_t)(b * CCH) + 32 + lh) * NPOS + i] = O2[r] * inv;
        out[((size_t)(b * CCH) + 48 + lh) * NPOS + i] = O3[r] * inv;
    }
}

// ---------------------------------------------------------------------------
extern "C" void kernel_launch(void* const* d_in, const int* in_sizes, int n_in,
                              void* d_out, int out_size, void* d_ws, size_t ws_size,
                              hipStream_t stream) {
    const float* x  = (const float*)d_in[0];
    const float* wq = (const float*)d_in[1];
    const float* bq = (const float*)d_in[2];
    const float* wk = (const float*)d_in[3];
    const float* bk = (const float*)d_in[4];
    const float* wv = (const float*)d_in[5];
    const float* bv = (const float*)d_in[6];
    float* out = (float*)d_out;

    // Workspace layout (bf16), all 16B aligned:
    __bf16* Wb   = (__bf16*)d_ws;                         // 3*64*576
    __bf16* Xcol = Wb   + (size_t)3 * CCH * K9;           // 8*2048*576
    __bf16* Qb   = Xcol + (size_t)BATCH * NPOS * K9;      // 8*2048*64 n-major
    __bf16* Kb   = Qb   + (size_t)BATCH * NPOS * CCH;     // 8*2048*64 n-major
    __bf16* Vb   = Kb   + (size_t)BATCH * NPOS * CCH;     // 8*64*2048 c-major
    // total ~25.4 MB

    {
        int total = 3 * CCH * K9;
        pack_weights<<<(total + 255) / 256, 256, 0, stream>>>(wq, wk, wv, Wb);
    }
    {
        size_t total = (size_t)BATCH * NPOS * K9;
        im2col_pack<<<(unsigned)((total + 255) / 256), 256, 0, stream>>>(x, Xcol);
    }
    {
        // 3 qkv * 8 b * 4 mtiles * 128 ntiles = 12288 waves, 4 waves/block
        qkv_gemm<<<12288 / 4, 128, 0, stream>>>(Wb, Xcol, bq, bk, bv, Qb, Kb, Vb);
    }
    {
        // 8 b * 128 qtiles = 1024 waves, 4 waves/block
        flash_attn<<<1024 / 4, 128, 0, stream>>>(Qb, Kb, Vb, out);
    }
}
